// RNNLayer_59708635349429
// MI455X (gfx1250) — matrix operand order repaired and checked by reading, not compiled, not run on hardware
//
#include <hip/hip_runtime.h>
#include <hip/hip_bf16.h>

#define INPUT  256
#define HIDDEN 512
#define OUTPUT 256
#define BATCH  64
#define TSTEPS 1024
#define MROWS  (BATCH * TSTEPS)

typedef __attribute__((ext_vector_type(16))) __bf16        v16bf;
typedef __attribute__((ext_vector_type(8)))  float         v8f;
typedef __attribute__((ext_vector_type(4)))  unsigned int  v4u;

// fp32 -> bf16 round-to-nearest-even
__device__ __forceinline__ unsigned short f2bf(float f) {
    unsigned int u = __float_as_uint(f);
    u += 0x7FFFu + ((u >> 16) & 1u);
    return (unsigned short)(u >> 16);
}

union FragU { v4u u4[2]; v16bf v; };

// ---------------------------------------------------------------------------
// Pack row-major fp32 W[K,N] into per-wave WMMA B fragments (bf16).
// Fragment f = ntile*(K/32) + ks occupies 256 dwords: [lane 0..31][dword 0..7].
// lane: n = ntile*16 + (lane&15), khalf = lane>>4.
// dword v holds bf16 pair for K = ks*32 + (v>>2)*16 + khalf*8 + 2*(v&3), +1.
// (Layout per CDNA5 ISA 16-bit B matrix striping.)
// ---------------------------------------------------------------------------
__global__ void pack_b_kernel(const float* __restrict__ W,
                              unsigned int* __restrict__ out,
                              int K, int N) {
    int tid = blockIdx.x * blockDim.x + threadIdx.x;
    int total = (K * N) >> 1;
    if (tid >= total) return;
    int f    = tid >> 8;
    int r    = tid & 255;
    int lane = r >> 3;
    int v    = r & 7;
    int ksteps = K >> 5;
    int ntile  = f / ksteps;
    int ks     = f - ntile * ksteps;
    int n  = ntile * 16 + (lane & 15);
    int k0 = ks * 32 + (v >> 2) * 16 + (lane >> 4) * 8 + 2 * (v & 3);
    unsigned int lo = f2bf(W[(long)k0 * N + n]);
    unsigned int hi = f2bf(W[(long)(k0 + 1) * N + n]);
    out[tid] = lo | (hi << 16);
}

// ---------------------------------------------------------------------------
// WMMA GEMM with fused bias: C[M,N] = A[M,K] @ W + bias.
// Compile-time: KSTEPS = K/32, A_BF16 = A element type (branch-free bodies).
// Block = 256 threads (8 waves), tile 64x64. wave w: wm=w&3 (16 rows),
// wn=w>>2 (2 column tiles of 16). M,N multiples of 64.
// ---------------------------------------------------------------------------
#define ATILE_S 40   // LDS row stride (elements); multiple of 8 -> 16B aligned rows
template <int KSTEPS, bool A_BF16>
__global__ __launch_bounds__(256) void wmma_gemm_bias(
                               const void* __restrict__ Aptr,
                               const v4u* __restrict__ Wfrag,
                               const float* __restrict__ bias,
                               float* __restrict__ Cout,
                               int N) {
    constexpr int K = KSTEPS * 32;
    __shared__ __align__(16) unsigned short shA[64 * ATILE_S];
    const int tid  = threadIdx.x;
    const int lane = tid & 31;
    const int w    = tid >> 5;
    const int wm   = w & 3;
    const int wn   = w >> 2;
    const long rowBase = (long)blockIdx.x * 64;
    const int  ntBase  = blockIdx.y * 4 + wn * 2;

    v8f c0 = {}; v8f c1 = {};

    const int ldRow = tid >> 2;        // 0..63
    const int ldCol = (tid & 3) * 8;   // 0,8,16,24

    const v4u* wf0 = Wfrag + ((long)(ntBase + 0) * KSTEPS) * 64 + lane * 2;
    const v4u* wf1 = Wfrag + ((long)(ntBase + 1) * KSTEPS) * 64 + lane * 2;
    const unsigned short* aBase = &shA[(wm * 16 + (lane & 15)) * ATILE_S + (lane >> 4) * 8];

    #pragma unroll
    for (int ks = 0; ks < KSTEPS; ++ks) {
        // ---- stage A tile [64 x 32] into LDS as bf16 ----
        {
            long gr = rowBase + ldRow;
            int  gc = ks * 32 + ldCol;
            __align__(16) unsigned short tmp[8];
            if (A_BF16) {
                const unsigned short* A = (const unsigned short*)Aptr;
                *(v4u*)tmp = *(const v4u*)(A + gr * K + gc);
            } else {
                const float* A = (const float*)Aptr;
                #pragma unroll
                for (int j = 0; j < 8; ++j) tmp[j] = f2bf(A[gr * K + gc + j]);
            }
            *(v4u*)&shA[ldRow * ATILE_S + ldCol] = *(const v4u*)tmp;
        }
        __syncthreads();

        FragU a;
        a.u4[0] = *(const v4u*)(aBase);
        a.u4[1] = *(const v4u*)(aBase + 16);
        FragU b0, b1;
        b0.u4[0] = wf0[ks * 64 + 0]; b0.u4[1] = wf0[ks * 64 + 1];
        b1.u4[0] = wf1[ks * 64 + 0]; b1.u4[1] = wf1[ks * 64 + 1];

        c0 = __builtin_amdgcn_wmma_f32_16x16x32_bf16(false, a.v, false, b0.v,
                                                     (short)0, c0, false, false);
        c1 = __builtin_amdgcn_wmma_f32_16x16x32_bf16(false, a.v, false, b1.v,
                                                     (short)0, c1, false, false);
        __syncthreads();
    }

    int col0 = ntBase * 16 + (lane & 15);
    int col1 = col0 + 16;
    int rsub = (lane >> 4) * 8;
    float bb0 = bias[col0], bb1 = bias[col1];
    #pragma unroll
    for (int r = 0; r < 8; ++r) {
        long row = rowBase + wm * 16 + rsub + r;
        Cout[row * N + col0] = c0[r] + bb0;
        Cout[row * N + col1] = c1[r] + bb1;
    }
}

// ---------------------------------------------------------------------------
// Sequential recurrence: h_t = tanh(h_{t-1} @ W_hh + xh_t), batch-parallel.
// Grid = 4 blocks x 512 threads (16 waves; launch_bounds(512,1) -> 4 waves/
// SIMD -> 256-VGPR budget on the 1024-VGPR/SIMD file). Block owns 16 batch
// rows (M=16). Wave w owns tiles 2w (W_hh strip register-resident, 128 VGPRs
// held across all 1024 steps) and 2w+1 (W_hh strip streamed from L2 each
// step). A 32-bit offset laundered through empty asm blocks LICM without
// destroying the pointer's global address space (pure global_load_b128,
// LOADcnt only). Shared A fragment feeds both WMMA chains (ILP=2).
// h double-buffered in LDS (stride 520). One barrier per timestep.
// ---------------------------------------------------------------------------
#define HSTR 520
__global__ __launch_bounds__(512, 1) void rnn_scan_kernel(
                                const float* __restrict__ xh,      // [B,T,512] fp32
                                const v4u*  __restrict__ Whh,      // packed frags
                                unsigned short* __restrict__ hs) { // [B,T,512] bf16
    __shared__ __align__(16) unsigned short hbuf[2][16 * HSTR];
    const int tid   = threadIdx.x;    // 0..511
    const int lane  = tid & 31;
    const int w     = tid >> 5;       // 0..15
    const int bbase = blockIdx.x * 16;

    for (int i = tid; i < 16 * HSTR; i += 512) hbuf[0][i] = 0;
    __syncthreads();

    const int m     = lane & 15;
    const int khalf = lane >> 4;
    const int col0  = w * 32 + m;     // tile 2w
    const int col1  = col0 + 16;      // tile 2w+1
    const int rsub  = khalf * 8;
    const int aoff  = m * HSTR + khalf * 8;

    // ---- resident weights: tile 2w, 16 fragments = 128 VGPRs ----
    FragU br[16];
    {
        const v4u* wf = Whh + ((long)(2 * w) * 16) * 64 + lane * 2;
        #pragma unroll
        for (int ks = 0; ks < 16; ++ks) {
            br[ks].u4[0] = wf[ks * 64 + 0];
            br[ks].u4[1] = wf[ks * 64 + 1];
        }
    }
    const v4u* wfs_base = Whh + ((long)(2 * w + 1) * 16) * 64 + lane * 2;

    for (int t = 0; t < TSTEPS; ++t) {
        const unsigned short* cur = hbuf[t & 1];
        unsigned short*       nxt = hbuf[(t & 1) ^ 1];
        v8f c0 = {}; v8f c1 = {};

        // opaque zero offset: defeats LICM (loads stay in-loop, L2 hits)
        // while keeping wfs in the global address space -> global_load_b128
        int dyn = 0;
        asm volatile("" : "+v"(dyn));
        const v4u* wfs = wfs_base + dyn;

        // prefetch next timestep's xh line (near locality -> WGP/GL2 scope)
        __builtin_prefetch(&xh[(((long)(bbase + rsub)) * TSTEPS + t + 1) * HIDDEN + col0], 0, 3);

        #pragma unroll
        for (int ks = 0; ks < 16; ++ks) {
            FragU a;
            const unsigned short* p = &cur[aoff + ks * 32];
            a.u4[0] = *(const v4u*)(p);
            a.u4[1] = *(const v4u*)(p + 16);
            FragU bs;
            bs.u4[0] = wfs[ks * 64 + 0];
            bs.u4[1] = wfs[ks * 64 + 1];
            c0 = __builtin_amdgcn_wmma_f32_16x16x32_bf16(false, a.v, false, br[ks].v,
                                                         (short)0, c0, false, false);
            c1 = __builtin_amdgcn_wmma_f32_16x16x32_bf16(false, a.v, false, bs.v,
                                                         (short)0, c1, false, false);
        }

        // tanh(c + xh_t) -> next LDS buffer + hs (bf16) for the output GEMM
        #pragma unroll
        for (int r = 0; r < 8; ++r) {
            int  row = rsub + r;
            long gi  = (((long)(bbase + row)) * TSTEPS + t) * HIDDEN;
            unsigned short u0 = f2bf(tanhf(c0[r] + xh[gi + col0]));
            unsigned short u1 = f2bf(tanhf(c1[r] + xh[gi + col1]));
            nxt[row * HSTR + col0] = u0;
            nxt[row * HSTR + col1] = u1;
            hs[gi + col0] = u0;
            hs[gi + col1] = u1;
        }
        __syncthreads();
    }
}

// ---------------------------------------------------------------------------
extern "C" void kernel_launch(void* const* d_in, const int* in_sizes, int n_in,
                              void* d_out, int out_size, void* d_ws, size_t ws_size,
                              hipStream_t stream) {
    const float* x    = (const float*)d_in[0];
    const float* W_xh = (const float*)d_in[1];
    const float* W_hh = (const float*)d_in[2];
    const float* b_h  = (const float*)d_in[3];
    const float* W_hy = (const float*)d_in[4];
    const float* b_y  = (const float*)d_in[5];
    float* out = (float*)d_out;

    // workspace layout (all regions fully rewritten every call):
    //   [0,256K)        packed W_xh bf16
    //   [256K,768K)     packed W_hh bf16
    //   [768K,1M)       packed W_hy bf16
    //   [1M, 1M+128M)   xh fp32  [B,T,512]
    //   then 64M        hs bf16  [B,T,512]
    char* ws = (char*)d_ws;
    unsigned int*   pWxh = (unsigned int*)(ws);
    unsigned int*   pWhh = (unsigned int*)(ws + (256 << 10));
    unsigned int*   pWhy = (unsigned int*)(ws + (768 << 10));
    float*          xh   = (float*)(ws + (1 << 20));
    unsigned short* hs   = (unsigned short*)(ws + (1 << 20) +
                                             (size_t)MROWS * HIDDEN * sizeof(float));

    pack_b_kernel<<<(INPUT  * HIDDEN / 2 + 255) / 256, 256, 0, stream>>>(W_xh, pWxh, INPUT,  HIDDEN);
    pack_b_kernel<<<(HIDDEN * HIDDEN / 2 + 255) / 256, 256, 0, stream>>>(W_hh, pWhh, HIDDEN, HIDDEN);
    pack_b_kernel<<<(HIDDEN * OUTPUT / 2 + 255) / 256, 256, 0, stream>>>(W_hy, pWhy, HIDDEN, OUTPUT);

    // xh = x @ W_xh + b_h   (M=65536, K=256, N=512)
    dim3 g1(MROWS / 64, HIDDEN / 64);
    wmma_gemm_bias<INPUT / 32, false><<<g1, 256, 0, stream>>>(
        x, (const v4u*)pWxh, b_h, xh, HIDDEN);

    // sequential scan over T, batch-parallel (4 blocks x 16 rows)
    rnn_scan_kernel<<<BATCH / 16, 512, 0, stream>>>(xh, (const v4u*)pWhh, hs);

    // out = hs @ W_hy + b_y  (M=65536, K=512, N=256)
    dim3 g3(MROWS / 64, OUTPUT / 64);
    wmma_gemm_bias<HIDDEN / 32, true><<<g3, 256, 0, stream>>>(
        hs, (const v4u*)pWhy, b_y, out, OUTPUT);
}
